// UserModule_3607772528806
// MI455X (gfx1250) — compile-verified, matrix operand order checked by reading, not run on hardware
//
#include <hip/hip_runtime.h>
#include <hip/hip_bf16.h>

typedef _Float16 v8h  __attribute__((ext_vector_type(8)));
typedef _Float16 v16h __attribute__((ext_vector_type(16)));
typedef float    v8f  __attribute__((ext_vector_type(8)));

#define BATCH  4096
#define EMB    16
#define NFEAT  32
#define NNZ    20
#define EMBOUT 512   // NFEAT * EMB
#define D1     1024
#define D2     512
#define D3     256

// ---------------- Stage 1: embedding gather + pool ----------------
// One thread per output element (b, f*16+e); 16 lanes share the same 20
// indices (broadcast load) and each lane reads one float of the 64B row
// -> fully coalesced 64B row fetches, L2-resident table (64MB < 192MB L2).
__global__ void emb_pool_kernel(const int* __restrict__ idx,
                                const float* __restrict__ table,
                                float* __restrict__ x) {
    int gid = blockIdx.x * blockDim.x + threadIdx.x;
    if (gid >= BATCH * EMBOUT) return;
    int b = gid >> 9;          // /512
    int c = gid & 511;
    int f = c >> 4;
    int e = c & 15;
    const int* ip = idx + (b * NFEAT + f) * NNZ;
    float s = 0.f;
#pragma unroll
    for (int l = 0; l < NNZ; ++l) {
        int t = ip[l];
        s += table[(size_t)t * EMB + e];
    }
    x[gid] = s;
}

// ---------------- Stage 2a: zero the BN accumulators ----------------
__global__ void zero_kernel(float* __restrict__ p, int n) {
    int i = blockIdx.x * blockDim.x + threadIdx.x;
    if (i < n) p[i] = 0.f;
}

// ---------------- Stage 2b: BN partial sums (row-coalesced) ----------------
// Each block reduces a 16-row x 512-col slab; thread t owns columns t and
// t+256 (consecutive addresses across the wave -> coalesced b32 loads),
// then commits one atomic f32 add per column per block.
__global__ void bn_partial_kernel(const float* __restrict__ x,
                                  float* __restrict__ sum,
                                  float* __restrict__ sumsq) {
    int r0 = blockIdx.x * 16;
    int c0 = threadIdx.x;
    int c1 = threadIdx.x + 256;
    float s0 = 0.f, q0 = 0.f, s1 = 0.f, q1 = 0.f;
#pragma unroll
    for (int r = 0; r < 16; ++r) {
        const float* row = x + (size_t)(r0 + r) * EMBOUT;
        float v0 = row[c0];
        float v1 = row[c1];
        s0 += v0; q0 += v0 * v0;
        s1 += v1; q1 += v1 * v1;
    }
    atomicAdd(&sum[c0], s0);
    atomicAdd(&sumsq[c0], q0);
    atomicAdd(&sum[c1], s1);
    atomicAdd(&sumsq[c1], q1);
}

// ---------------- Stage 3: BN apply + f16 quantize ----------------
__global__ void bn_apply_kernel(const float* __restrict__ x,
                                const float* __restrict__ sum,
                                const float* __restrict__ sumsq,
                                const float* __restrict__ gamma,
                                const float* __restrict__ beta,
                                _Float16* __restrict__ xh) {
    int gid = blockIdx.x * blockDim.x + threadIdx.x;
    if (gid >= BATCH * EMBOUT) return;
    int c = gid & 511;
    const float inv = 1.0f / BATCH;
    float m   = sum[c] * inv;
    float var = sumsq[c] * inv - m * m;
    float v = (x[gid] - m) * rsqrtf(var + 1e-5f) * gamma[c] + beta[c];
    xh[gid] = (_Float16)v;
}

// ---------------- Weight prep: fp32 [K,N] -> f16 [N,K] ----------------
__global__ void wconv_kernel(const float* __restrict__ W,
                             _Float16* __restrict__ Wt, int K, int N) {
    int gid = blockIdx.x * blockDim.x + threadIdx.x;
    if (gid >= K * N) return;
    int n = gid % N;
    int k = gid / N;               // coalesced read of W row-major
    Wt[(size_t)n * K + k] = (_Float16)W[gid];
}

// ---- 16-bit 16x32 WMMA fragment load (ISA 7.12.2 layout) ----
// lanes 0-15 : row = lane,    halves 0..7 = K 0..7,  8..15 = K 16..23
// lanes 16-31: row = lane-16, halves 0..7 = K 8..15, 8..15 = K 24..31
__device__ __forceinline__ v16h ldfrag(const _Float16* p) {
    v8h x0 = *(const v8h*)p;
    v8h x1 = *(const v8h*)(p + 16);
    return __builtin_shufflevector(x0, x1, 0,1,2,3,4,5,6,7,8,9,10,11,12,13,14,15);
}

__device__ __forceinline__ v8f wmma_f16(v16h a, v16h b, v8f c) {
    return __builtin_amdgcn_wmma_f32_16x16x32_f16(
        /*neg_a=*/false, a, /*neg_b=*/false, b,
        /*c_mod=*/(short)0, c, /*reuse_a=*/false, /*reuse_b=*/false);
}

// ---------------- WMMA GEMM: C[M,N] = A[M,K] * Bt[N,K]^T ----------------
// Register-blocked: each wave computes a 32x32 output block (2x2 grid of
// 16x16 WMMA tiles). Per 32-wide K step: 2 A frags + 2 B frags feed 4
// v_wmma_f32_16x16x32_f16 -> 16 FLOP/byte from L2 (2x the 1-tile version).
// Fused bias(+ReLU) epilogue with f16 requant for the next layer.
__global__ void __launch_bounds__(128)
gemm_wmma_kernel(const _Float16* __restrict__ A,
                 const _Float16* __restrict__ Bt,
                 const float* __restrict__ bias,
                 _Float16* __restrict__ outH,   // f16 activation (or null)
                 float* __restrict__ outF,      // fp32 output (or null)
                 int M, int N, int K, int doRelu) {
    int wave = blockIdx.x * (blockDim.x >> 5) + (threadIdx.x >> 5);
    int lane = threadIdx.x & 31;
    int nblk = N >> 5;              // 32-wide N blocks
    int mb   = wave / nblk;
    int nb   = wave % nblk;
    if (mb >= (M >> 5)) return;

    int half_sel = lane >> 4;       // selects K-offset 0 vs 8 within the frag
    int l15      = lane & 15;

    const _Float16* a0p = A  + (size_t)(mb * 32 + l15) * K + half_sel * 8;
    const _Float16* a1p = a0p + (size_t)16 * K;
    const _Float16* b0p = Bt + (size_t)(nb * 32 + l15) * K + half_sel * 8;
    const _Float16* b1p = b0p + (size_t)16 * K;

    v8f c00 = {}, c01 = {}, c10 = {}, c11 = {};
    for (int kk = 0; kk < K; kk += 32) {
        v16h a0 = ldfrag(a0p + kk);
        v16h a1 = ldfrag(a1p + kk);
        v16h b0 = ldfrag(b0p + kk);
        v16h b1 = ldfrag(b1p + kk);
        c00 = wmma_f16(a0, b0, c00);
        c01 = wmma_f16(a0, b1, c01);
        c10 = wmma_f16(a1, b0, c10);
        c11 = wmma_f16(a1, b1, c11);
    }

    // C/D layout: VGPR r -> row r (lanes 0-15) / r+8 (lanes 16-31), col = lane&15
    int col0 = nb * 32 + l15;
    float bc0 = bias[col0];
    float bc1 = bias[col0 + 16];
#pragma unroll
    for (int mi = 0; mi < 2; ++mi) {
        v8f cm0 = mi ? c10 : c00;
        v8f cm1 = mi ? c11 : c01;
#pragma unroll
        for (int r = 0; r < 8; ++r) {
            int row = mb * 32 + mi * 16 + r + half_sel * 8;
            float v0 = cm0[r] + bc0;
            float v1 = cm1[r] + bc1;
            if (doRelu) { v0 = fmaxf(v0, 0.f); v1 = fmaxf(v1, 0.f); }
            size_t o = (size_t)row * N + col0;
            if (outH) { outH[o] = (_Float16)v0; outH[o + 16] = (_Float16)v1; }
            if (outF) { outF[o] = v0;           outF[o + 16] = v1; }
        }
    }
}

// ---------------- Host-side launcher ----------------
extern "C" void kernel_launch(void* const* d_in, const int* in_sizes, int n_in,
                              void* d_out, int out_size, void* d_ws, size_t ws_size,
                              hipStream_t stream) {
    const int*   indices = (const int*)  d_in[0];
    const float* table   = (const float*)d_in[1];
    const float* gamma   = (const float*)d_in[2];
    const float* beta    = (const float*)d_in[3];
    const float* W1      = (const float*)d_in[4];
    const float* b1      = (const float*)d_in[5];
    const float* W2      = (const float*)d_in[6];
    const float* b2      = (const float*)d_in[7];
    const float* W3      = (const float*)d_in[8];
    const float* b3      = (const float*)d_in[9];
    float* out = (float*)d_out;

    // Carve workspace (all sizes multiples of 256B)
    char* p = (char*)d_ws;
    float*    x     = (float*)p;    p += (size_t)BATCH * EMBOUT * sizeof(float);    // 8 MB
    _Float16* xh    = (_Float16*)p; p += (size_t)BATCH * EMBOUT * sizeof(_Float16); // 4 MB
    _Float16* h1    = (_Float16*)p; p += (size_t)BATCH * D1     * sizeof(_Float16); // 8 MB
    _Float16* h2    = (_Float16*)p; p += (size_t)BATCH * D2     * sizeof(_Float16); // 4 MB
    _Float16* W1t   = (_Float16*)p; p += (size_t)D1 * EMBOUT    * sizeof(_Float16); // 1 MB
    _Float16* W2t   = (_Float16*)p; p += (size_t)D2 * D1        * sizeof(_Float16); // 1 MB
    _Float16* W3t   = (_Float16*)p; p += (size_t)D3 * D2        * sizeof(_Float16); // 0.25 MB
    float*    sum   = (float*)p;    p += 512 * sizeof(float);
    float*    sumsq = (float*)p;    p += 512 * sizeof(float);
    (void)ws_size; (void)in_sizes; (void)n_in; (void)out_size;

    // 1) gather + pool
    emb_pool_kernel<<<(BATCH * EMBOUT) / 256, 256, 0, stream>>>(indices, table, x);
    // 2) BN statistics: zero accumulators, then coalesced partial reduce
    zero_kernel<<<4, 256, 0, stream>>>(sum, 1024);   // zeros sum AND sumsq (contiguous)
    bn_partial_kernel<<<BATCH / 16, 256, 0, stream>>>(x, sum, sumsq);
    // 3) BN apply + f16 cast (mean/rstd reconstructed inline)
    bn_apply_kernel<<<(BATCH * EMBOUT) / 256, 256, 0, stream>>>(x, sum, sumsq, gamma, beta, xh);
    // Weight prep (independent of 1-3; scheduler may overlap)
    wconv_kernel<<<(EMBOUT * D1) / 256, 256, 0, stream>>>(W1, W1t, EMBOUT, D1);
    wconv_kernel<<<(D1 * D2) / 256, 256, 0, stream>>>(W2, W2t, D1, D2);
    wconv_kernel<<<(D2 * D3) / 256, 256, 0, stream>>>(W3, W3t, D2, D3);

    // 4) MLP: 512 -> 1024 (ReLU), 1024 -> 512 (ReLU), 512 -> 256
    {
        int waves = (BATCH / 32) * (D1 / 32);   // 4096 waves
        gemm_wmma_kernel<<<waves / 4, 128, 0, stream>>>(xh, W1t, b1, h1, nullptr,
                                                        BATCH, D1, EMBOUT, 1);
    }
    {
        int waves = (BATCH / 32) * (D2 / 32);   // 2048 waves
        gemm_wmma_kernel<<<waves / 4, 128, 0, stream>>>(h1, W2t, b2, h2, nullptr,
                                                        BATCH, D2, D1, 1);
    }
    {
        int waves = (BATCH / 32) * (D3 / 32);   // 1024 waves
        gemm_wmma_kernel<<<waves / 4, 128, 0, stream>>>(h2, W3t, b3, nullptr, out,
                                                        BATCH, D3, D2, 0);
    }
}
